// Seq2SeqEncoderDecoder_1142461301362
// MI455X (gfx1250) — matrix operand order repaired and checked by reading, not compiled, not run on hardware
//
#include <hip/hip_runtime.h>
#include <hip/hip_bf16.h>

// ---------------- constants ----------------
enum { VV = 50000, EE = 300, EP = 320, HH = 256, NB = 64, LSs = 100, LTt = 20 };
enum { G4H = 4 * HH };           // 1024
// d_out layout: logits (LT,N,V) then scores (LT,N,LS,1)

typedef __attribute__((ext_vector_type(16))) _Float16 v16h;
typedef __attribute__((ext_vector_type(8)))  float    v8f;

// ---------------- generic WMMA GEMM (software-pipelined) ----------------
// C[r][c] = act( sum_k A[r][k]*Bh[c][k] + bias[c] + Cin[r][c] )
// A : (M x Kp) fp32 row-major, lda (multiple of 32), zero-padded K tail
// Bh: (Ncp x Kp) f16 row-major, fully zero-padded (no bound checks needed)
// Block: 256 threads = 8 waves, tile 64(M) x 128(N); wave tile 16 x 64.
__global__ __launch_bounds__(256)
void k_gemm(const float* __restrict__ A, int lda,
            const _Float16* __restrict__ Bh, int ldb,   // ldb == Kp
            const float* __restrict__ bias,
            const float* __restrict__ Cin, int ldcin,
            float* __restrict__ Out, long long ldout,
            int M, int Nc, int Kp,
            int act,                  // 0 = none, 1 = tanh
            int remapLT, int remapN)  // if remapLT>0: row r -> (r%LT)*remapN + r/LT
{
    const int lane = threadIdx.x & 31;
    const int wave = threadIdx.x >> 5;
    const int row0 = blockIdx.y * 64 + (wave & 3) * 16;
    const int col0 = blockIdx.x * 128 + (wave >> 2) * 64;
    if (row0 >= M) return;

    v8f acc0 = {}, acc1 = {}, acc2 = {}, acc3 = {};

    const int rA    = row0 + (lane & 15);
    const int kselA = (lane >> 4) * 16;   // A frag: lanes 0-15 -> K 0..15, 16-31 -> 16..31
    const int cB    = col0 + (lane & 15);
    const int kselB = (lane >> 4) * 16;   // B frag: lanes 0-15 -> K 0..15, 16-31 -> 16..31

    const float*    arow = A + (size_t)rA * lda + kselA;
    const _Float16* pb0  = Bh + (size_t)(cB +  0) * ldb + kselB;
    const _Float16* pb1  = Bh + (size_t)(cB + 16) * ldb + kselB;
    const _Float16* pb2  = Bh + (size_t)(cB + 32) * ldb + kselB;
    const _Float16* pb3  = Bh + (size_t)(cB + 48) * ldb + kselB;

    // ---- prologue: load chunk k=0 ----
    const float4* ap = (const float4*)arow;
    float4 ra0 = ap[0], ra1 = ap[1], ra2 = ap[2], ra3 = ap[3];
    v16h  rb0 = *(const v16h*)pb0;
    v16h  rb1 = *(const v16h*)pb1;
    v16h  rb2 = *(const v16h*)pb2;
    v16h  rb3 = *(const v16h*)pb3;

    for (int k0 = 0; k0 < Kp; k0 += 32) {
        // current chunk (already in registers)
        float4 ca0 = ra0, ca1 = ra1, ca2 = ra2, ca3 = ra3;
        v16h  cb0 = rb0, cb1 = rb1, cb2 = rb2, cb3 = rb3;

        // prefetch next chunk (wraps to 0 on last iter; branchless)
        int kn = (k0 + 32 < Kp) ? (k0 + 32) : 0;
        const float4* apn = (const float4*)(arow + kn);
        ra0 = apn[0]; ra1 = apn[1]; ra2 = apn[2]; ra3 = apn[3];
        rb0 = *(const v16h*)(pb0 + kn);
        rb1 = *(const v16h*)(pb1 + kn);
        rb2 = *(const v16h*)(pb2 + kn);
        rb3 = *(const v16h*)(pb3 + kn);

        // cvt current A -> f16 fragment
        float av[16];
        *(float4*)(av +  0) = ca0;
        *(float4*)(av +  4) = ca1;
        *(float4*)(av +  8) = ca2;
        *(float4*)(av + 12) = ca3;
        v16h a;
#pragma unroll
        for (int i = 0; i < 16; ++i) a[i] = (_Float16)av[i];

        acc0 = __builtin_amdgcn_wmma_f32_16x16x32_f16(false, a, false, cb0, (short)0, acc0, false, false);
        acc1 = __builtin_amdgcn_wmma_f32_16x16x32_f16(false, a, false, cb1, (short)0, acc1, false, false);
        acc2 = __builtin_amdgcn_wmma_f32_16x16x32_f16(false, a, false, cb2, (short)0, acc2, false, false);
        acc3 = __builtin_amdgcn_wmma_f32_16x16x32_f16(false, a, false, cb3, (short)0, acc3, false, false);
    }

    // C layout: acc[v], lane -> row = row0 + v + (lane>>4)*8, col = tile + (lane&15)
    const int rbase = row0 + (lane >> 4) * 8;
#pragma unroll
    for (int j = 0; j < 4; ++j) {
        const v8f acc = (j == 0) ? acc0 : (j == 1) ? acc1 : (j == 2) ? acc2 : acc3;
        int c = col0 + j * 16 + (lane & 15);
        if (c >= Nc) continue;
        float bv = bias ? bias[c] : 0.f;
#pragma unroll
        for (int v = 0; v < 8; ++v) {
            int r = rbase + v;
            float val = acc[v] + bv;
            if (Cin) val += Cin[(long long)r * ldcin + c];
            if (act) val = tanhf(val);
            if (remapLT > 0) {
                long long orow = (long long)(r % remapLT) * remapN + (r / remapLT);
                __builtin_nontemporal_store(val, &Out[orow * ldout + c]);
            } else {
                Out[(long long)r * ldout + c] = val;
            }
        }
    }
}

// ---------------- weight fp32 -> padded f16 ----------------
__global__ void k_wconv(const float* __restrict__ W, _Float16* __restrict__ Wh,
                        int Nc, int K, int Kp, long long total)
{
    long long i = (long long)blockIdx.x * blockDim.x + threadIdx.x;
    if (i >= total) return;
    int r = (int)(i / Kp), k = (int)(i % Kp);
    float v = 0.f;
    if (r < Nc && k < K) v = W[(long long)r * K + k];
    Wh[i] = (_Float16)v;
}

// ---------------- pointwise LSTM gates ----------------
__global__ void k_lstm_point(const float* __restrict__ G,
                             float* __restrict__ c, float* __restrict__ h,
                             float* __restrict__ out,
                             int tpos, int L, int ldout, int coloff)
{
    int idx = blockIdx.x * blockDim.x + threadIdx.x;   // 0 .. 64*256
    if (idx >= NB * HH) return;
    int n = idx / HH, j = idx % HH;
    const float* g = G + (long long)n * G4H;
    float gi = g[j], gf = g[HH + j], gg = g[2 * HH + j], go = g[3 * HH + j];
    float si = 1.f / (1.f + __expf(-gi));
    float sf = 1.f / (1.f + __expf(-gf));
    float so = 1.f / (1.f + __expf(-go));
    float cc = sf * c[idx] + si * tanhf(gg);
    float hh = so * tanhf(cc);
    c[idx] = cc;
    h[idx] = hh;
    out[((long long)n * L + tpos) * ldout + coloff + j] = hh;
}

// ---------------- embedding gather (stride EP, zero pad) ----------------
__global__ void k_embed(const float* __restrict__ emb, const int* __restrict__ tok,
                        float* __restrict__ out, long long total)
{
    long long i = (long long)blockIdx.x * blockDim.x + threadIdx.x;
    if (i >= total) return;
    int r = (int)(i / EP), e = (int)(i % EP);
    out[i] = (e < EE) ? emb[(long long)tok[r] * EE + e] : 0.f;
}

__global__ void k_zero(float* __restrict__ p, long long n)
{
    long long i = (long long)blockIdx.x * blockDim.x + threadIdx.x;
    if (i < n) p[i] = 0.f;
}

// ---------------- attention ----------------
__global__ void k_attn_logits(const float* __restrict__ enc, const float* __restrict__ q,
                              float* __restrict__ alog)
{
    int i = blockIdx.x * blockDim.x + threadIdx.x;
    if (i >= NB * LTt * LSs) return;
    int s = i % LSs;
    int t = (i / LSs) % LTt;
    int n = i / (LSs * LTt);
    const float* ep = enc + ((long long)n * LSs + s) * (2 * HH);
    const float* qp = q + ((long long)n * LTt + t) * (2 * HH);
    float acc = 0.f;
    for (int e = 0; e < 2 * HH; ++e) acc += ep[e] * qp[e];
    alog[i] = acc;
}

__global__ void k_softmax(const float* __restrict__ alog, float* __restrict__ scores)
{
    int lane = threadIdx.x & 31;
    int wave = threadIdx.x >> 5;
    int row = blockIdx.x * 8 + wave;             // row = n*LT + t
    if (row >= NB * LTt) return;
    int n = row / LTt, t = row % LTt;
    const float* a = alog + (long long)row * LSs;
    float m = -1e30f;
    for (int s = lane; s < LSs; s += 32) m = fmaxf(m, a[s]);
    for (int o = 16; o; o >>= 1) m = fmaxf(m, __shfl_xor(m, o, 32));
    float sum = 0.f;
    for (int s = lane; s < LSs; s += 32) sum += __expf(a[s] - m);
    for (int o = 16; o; o >>= 1) sum += __shfl_xor(sum, o, 32);
    float inv = 1.f / sum;
    float* op = scores + ((long long)t * NB + n) * LSs;
    for (int s = lane; s < LSs; s += 32) op[s] = __expf(a[s] - m) * inv;
}

__global__ void k_ctx(const float* __restrict__ scores, const float* __restrict__ enc,
                      float* __restrict__ ctx)
{
    int i = blockIdx.x * blockDim.x + threadIdx.x;    // NB*LTt*512
    if (i >= NB * LTt * 2 * HH) return;
    int e = i % (2 * HH);
    int row = i / (2 * HH);
    int t = row % LTt, n = row / LTt;
    const float* sp = scores + ((long long)t * NB + n) * LSs;
    const float* ep = enc + (long long)n * LSs * (2 * HH) + e;
    float acc = 0.f;
    for (int s = 0; s < LSs; ++s) acc += sp[s] * ep[(long long)s * (2 * HH)];
    ctx[i] = acc;
}

__global__ void k_cat(const float* __restrict__ dec, const float* __restrict__ ctx,
                      float* __restrict__ cat)
{
    int i = blockIdx.x * blockDim.x + threadIdx.x;    // 1280*768
    if (i >= NB * LTt * 3 * HH) return;
    int c = i % (3 * HH);
    int r = i / (3 * HH);
    cat[i] = (c < HH) ? dec[(long long)r * HH + c]
                      : ctx[(long long)r * (2 * HH) + (c - HH)];
}

// ---------------- host helpers ----------------
static inline void gemm(hipStream_t s, const float* A, int lda,
                        const _Float16* Bh, int ldb,
                        const float* bias, const float* Cin, int ldcin,
                        float* Out, long long ldout, int M, int Nc, int Kp,
                        int act = 0, int remapLT = 0, int remapN = 0)
{
    dim3 grid((Nc + 127) / 128, (M + 63) / 64);
    k_gemm<<<grid, 256, 0, s>>>(A, lda, Bh, ldb, bias, Cin, ldcin, Out, ldout,
                                M, Nc, Kp, act, remapLT, remapN);
}

static inline void wconv(hipStream_t s, const float* W, _Float16* Wh,
                         int Nc, int K, int Ncp, int Kp)
{
    long long total = (long long)Ncp * Kp;
    k_wconv<<<(int)((total + 255) / 256), 256, 0, s>>>(W, Wh, Nc, K, Kp, total);
}

extern "C" void kernel_launch(void* const* d_in, const int* in_sizes, int n_in,
                              void* d_out, int out_size, void* d_ws, size_t ws_size,
                              hipStream_t stream)
{
    (void)in_sizes; (void)n_in; (void)out_size; (void)ws_size;
    const int*   source   = (const int*)  d_in[0];
    const int*   target   = (const int*)  d_in[1];
    const float* emb      = (const float*)d_in[2];
    const float* Wih_l0f  = (const float*)d_in[3];
    const float* Whh_l0f  = (const float*)d_in[4];
    const float* b_l0f    = (const float*)d_in[5];
    const float* Wih_l0b  = (const float*)d_in[6];
    const float* Whh_l0b  = (const float*)d_in[7];
    const float* b_l0b    = (const float*)d_in[8];
    const float* Wih_l1f  = (const float*)d_in[9];
    const float* Whh_l1f  = (const float*)d_in[10];
    const float* b_l1f    = (const float*)d_in[11];
    const float* Wih_l1b  = (const float*)d_in[12];
    const float* Whh_l1b  = (const float*)d_in[13];
    const float* b_l1b    = (const float*)d_in[14];
    const float* dWih0    = (const float*)d_in[15];
    const float* dWhh0    = (const float*)d_in[16];
    const float* db0      = (const float*)d_in[17];
    const float* dWih1    = (const float*)d_in[18];
    const float* dWhh1    = (const float*)d_in[19];
    const float* db1      = (const float*)d_in[20];
    const float* attn_W   = (const float*)d_in[21];
    const float* attn_b   = (const float*)d_in[22];
    const float* mlp_W1   = (const float*)d_in[23];
    const float* mlp_b1   = (const float*)d_in[24];
    const float* mlp_W2   = (const float*)d_in[25];
    const float* mlp_b2   = (const float*)d_in[26];
    float* out = (float*)d_out;

    // ---- workspace layout: fp32 region ----
    float* ws     = (float*)d_ws;
    float* X_SRC  = ws;                         // 6400*320   = 2,048,000
    float* X_TGT  = X_SRC  + 2048000;           // 1280*320   =   409,600
    float* GX_A   = X_TGT  + 409600;            // 6400*1024  = 6,553,600
    float* GX_B   = GX_A   + 6553600;           // 6400*1024  = 6,553,600
    float* L0OUT  = GX_B   + 6553600;           // 6400*512   = 3,276,800
    float* ENCOUT = L0OUT  + 3276800;           // 6400*512   = 3,276,800
    float* G_STEP = ENCOUT + 3276800;           // 64*1024    =    65,536
    float* HC     = G_STEP + 65536;             // 12*16384   =   196,608
    float* GXD0   = HC     + 196608;            // 1280*1024  = 1,310,720
    float* GXD1   = GXD0   + 1310720;           // 1280*1024  = 1,310,720
    float* D0     = GXD1   + 1310720;           // 1280*256   =   327,680
    float* DECOUT = D0     + 327680;            // 1280*256   =   327,680
    float* Q      = DECOUT + 327680;            // 1280*512   =   655,360
    float* ALOG   = Q      + 655360;            // 64*20*100  =   128,000
    float* CTX    = ALOG   + 128000;            // 1280*512   =   655,360
    float* CAT    = CTX    + 655360;            // 1280*768   =   983,040
    float* HDD    = CAT    + 983040;            // 1280*256   =   327,680
    float* FEND   = HDD    + 327680;

    // ---- f16 weight region ----
    _Float16* W16 = (_Float16*)FEND;
    _Float16* F_Wih_l0f = W16;                  // 1024x320
    _Float16* F_Wih_l0b = F_Wih_l0f + 327680;   // 1024x320
    _Float16* F_Whh_l0f = F_Wih_l0b + 327680;   // 1024x256
    _Float16* F_Whh_l0b = F_Whh_l0f + 262144;
    _Float16* F_Wih_l1f = F_Whh_l0b + 262144;   // 1024x512
    _Float16* F_Wih_l1b = F_Wih_l1f + 524288;
    _Float16* F_Whh_l1f = F_Wih_l1b + 524288;   // 1024x256
    _Float16* F_Whh_l1b = F_Whh_l1f + 262144;
    _Float16* F_dWih0   = F_Whh_l1b + 262144;   // 1024x320
    _Float16* F_dWhh0   = F_dWih0   + 327680;   // 1024x256
    _Float16* F_dWih1   = F_dWhh0   + 262144;   // 1024x256
    _Float16* F_dWhh1   = F_dWih1   + 262144;   // 1024x256
    _Float16* F_attnW   = F_dWhh1   + 262144;   // 512x256
    _Float16* F_mlpW1   = F_attnW   + 131072;   // 256x768
    _Float16* F_mlpW2   = F_mlpW1   + 196608;   // 50048x256

    float* h_l0f = HC;            float* c_l0f = HC + 16384;
    float* h_l0b = HC + 32768;    float* c_l0b = HC + 49152;
    float* h_l1f = HC + 65536;    float* c_l1f = HC + 81920;
    float* h_l1b = HC + 98304;    float* c_l1b = HC + 114688;
    float* h_d0  = HC + 131072;   float* c_d0  = HC + 147456;
    float* h_d1  = HC + 163840;   float* c_d1  = HC + 180224;

    // ---- weight conversions (fp32 -> padded f16) ----
    wconv(stream, Wih_l0f, F_Wih_l0f, G4H, EE, G4H, EP);
    wconv(stream, Wih_l0b, F_Wih_l0b, G4H, EE, G4H, EP);
    wconv(stream, Whh_l0f, F_Whh_l0f, G4H, HH, G4H, HH);
    wconv(stream, Whh_l0b, F_Whh_l0b, G4H, HH, G4H, HH);
    wconv(stream, Wih_l1f, F_Wih_l1f, G4H, 2 * HH, G4H, 2 * HH);
    wconv(stream, Wih_l1b, F_Wih_l1b, G4H, 2 * HH, G4H, 2 * HH);
    wconv(stream, Whh_l1f, F_Whh_l1f, G4H, HH, G4H, HH);
    wconv(stream, Whh_l1b, F_Whh_l1b, G4H, HH, G4H, HH);
    wconv(stream, dWih0,   F_dWih0,   G4H, EE, G4H, EP);
    wconv(stream, dWhh0,   F_dWhh0,   G4H, HH, G4H, HH);
    wconv(stream, dWih1,   F_dWih1,   G4H, HH, G4H, HH);
    wconv(stream, dWhh1,   F_dWhh1,   G4H, HH, G4H, HH);
    wconv(stream, attn_W,  F_attnW,   2 * HH, HH, 2 * HH, HH);
    wconv(stream, mlp_W1,  F_mlpW1,   HH, 3 * HH, HH, 3 * HH);
    wconv(stream, mlp_W2,  F_mlpW2,   VV, HH, 50048, HH);

    const long long nSrc = (long long)NB * LSs * EP;
    const long long nTgt = (long long)NB * LTt * EP;

    // ---- embeddings (stride EP, zero-padded) ----
    k_embed<<<(int)((nSrc + 255) / 256), 256, 0, stream>>>(emb, source, X_SRC, nSrc);
    k_embed<<<(int)((nTgt + 255) / 256), 256, 0, stream>>>(emb, target, X_TGT, nTgt);

    // ---- zero h/c states ----
    k_zero<<<(196608 + 255) / 256, 256, 0, stream>>>(HC, 196608);

    // ---- encoder layer 0: hoisted input transforms ----
    gemm(stream, X_SRC, EP, F_Wih_l0f, EP, b_l0f, nullptr, 0, GX_A, G4H, NB * LSs, G4H, EP);
    gemm(stream, X_SRC, EP, F_Wih_l0b, EP, b_l0b, nullptr, 0, GX_B, G4H, NB * LSs, G4H, EP);

    for (int t = 0; t < LSs; ++t) {
        gemm(stream, h_l0f, HH, F_Whh_l0f, HH, nullptr, GX_A + (long long)t * G4H, LSs * G4H,
             G_STEP, G4H, NB, G4H, HH);
        k_lstm_point<<<64, 256, 0, stream>>>(G_STEP, c_l0f, h_l0f, L0OUT, t, LSs, 2 * HH, 0);
    }
    for (int ts = 0; ts < LSs; ++ts) {
        int sp = LSs - 1 - ts;
        gemm(stream, h_l0b, HH, F_Whh_l0b, HH, nullptr, GX_B + (long long)sp * G4H, LSs * G4H,
             G_STEP, G4H, NB, G4H, HH);
        k_lstm_point<<<64, 256, 0, stream>>>(G_STEP, c_l0b, h_l0b, L0OUT, sp, LSs, 2 * HH, HH);
    }

    // ---- encoder layer 1 ----
    gemm(stream, L0OUT, 2 * HH, F_Wih_l1f, 2 * HH, b_l1f, nullptr, 0, GX_A, G4H, NB * LSs, G4H, 2 * HH);
    gemm(stream, L0OUT, 2 * HH, F_Wih_l1b, 2 * HH, b_l1b, nullptr, 0, GX_B, G4H, NB * LSs, G4H, 2 * HH);
    for (int t = 0; t < LSs; ++t) {
        gemm(stream, h_l1f, HH, F_Whh_l1f, HH, nullptr, GX_A + (long long)t * G4H, LSs * G4H,
             G_STEP, G4H, NB, G4H, HH);
        k_lstm_point<<<64, 256, 0, stream>>>(G_STEP, c_l1f, h_l1f, ENCOUT, t, LSs, 2 * HH, 0);
    }
    for (int ts = 0; ts < LSs; ++ts) {
        int sp = LSs - 1 - ts;
        gemm(stream, h_l1b, HH, F_Whh_l1b, HH, nullptr, GX_B + (long long)sp * G4H, LSs * G4H,
             G_STEP, G4H, NB, G4H, HH);
        k_lstm_point<<<64, 256, 0, stream>>>(G_STEP, c_l1b, h_l1b, ENCOUT, sp, LSs, 2 * HH, HH);
    }

    // ---- decoder init: layer0 <- enc l0 fwd finals, layer1 <- enc l0 bwd finals ----
    size_t hcBytes = (size_t)NB * HH * sizeof(float);
    hipMemcpyAsync(h_d0, h_l0f, hcBytes, hipMemcpyDeviceToDevice, stream);
    hipMemcpyAsync(c_d0, c_l0f, hcBytes, hipMemcpyDeviceToDevice, stream);
    hipMemcpyAsync(h_d1, h_l0b, hcBytes, hipMemcpyDeviceToDevice, stream);
    hipMemcpyAsync(c_d1, c_l0b, hcBytes, hipMemcpyDeviceToDevice, stream);

    // ---- decoder layer 0 ----
    gemm(stream, X_TGT, EP, F_dWih0, EP, db0, nullptr, 0, GXD0, G4H, NB * LTt, G4H, EP);
    for (int t = 0; t < LTt; ++t) {
        gemm(stream, h_d0, HH, F_dWhh0, HH, nullptr, GXD0 + (long long)t * G4H, LTt * G4H,
             G_STEP, G4H, NB, G4H, HH);
        k_lstm_point<<<64, 256, 0, stream>>>(G_STEP, c_d0, h_d0, D0, t, LTt, HH, 0);
    }
    // ---- decoder layer 1 ----
    gemm(stream, D0, HH, F_dWih1, HH, db1, nullptr, 0, GXD1, G4H, NB * LTt, G4H, HH);
    for (int t = 0; t < LTt; ++t) {
        gemm(stream, h_d1, HH, F_dWhh1, HH, nullptr, GXD1 + (long long)t * G4H, LTt * G4H,
             G_STEP, G4H, NB, G4H, HH);
        k_lstm_point<<<64, 256, 0, stream>>>(G_STEP, c_d1, h_d1, DECOUT, t, LTt, HH, 0);
    }

    // ---- attention ----
    gemm(stream, DECOUT, HH, F_attnW, HH, attn_b, nullptr, 0, Q, 2 * HH, NB * LTt, 2 * HH, HH);
    {
        int tot = NB * LTt * LSs;
        k_attn_logits<<<(tot + 255) / 256, 256, 0, stream>>>(ENCOUT, Q, ALOG);
    }
    float* scores = out + (long long)LTt * NB * VV;
    k_softmax<<<(NB * LTt + 7) / 8, 256, 0, stream>>>(ALOG, scores);
    {
        int tot = NB * LTt * 2 * HH;
        k_ctx<<<(tot + 255) / 256, 256, 0, stream>>>(scores, ENCOUT, CTX);
    }
    {
        int tot = NB * LTt * 3 * HH;
        k_cat<<<(tot + 255) / 256, 256, 0, stream>>>(DECOUT, CTX, CAT);
    }

    // ---- output MLP ----
    gemm(stream, CAT, 3 * HH, F_mlpW1, 3 * HH, mlp_b1, nullptr, 0, HDD, HH, NB * LTt, HH, 3 * HH,
         /*act=*/1);
    // logits -> d_out with (n,t) -> (t,n) row remap, non-temporal stores
    gemm(stream, HDD, HH, F_mlpW2, HH, mlp_b2, nullptr, 0, out, VV, NB * LTt, VV, HH,
         /*act=*/0, /*remapLT=*/LTt, /*remapN=*/NB);
}
